// Ernie4_5_VLSparseMoeBlock_63720134803582
// MI455X (gfx1250) — compile-verified, matrix operand order checked by reading, not run on hardware
//
#include <hip/hip_runtime.h>

// ---------------------------------------------------------------------------
// Ernie4.5 sparse MoE block for MI455X (gfx1250), wave32 + WMMA bf16 + TDM.
// Memory-bound design: stream 805MB of expert weights exactly once,
// weight-stationary bf16 tiles in LDS, v_wmma_f32_16x16x32_bf16 inner loops,
// Tensor Data Mover staging for the down-proj weights.
// ---------------------------------------------------------------------------

typedef __attribute__((ext_vector_type(16))) __bf16 v16bf;
typedef __attribute__((ext_vector_type(8)))  float  v8f;
typedef unsigned int v4u __attribute__((ext_vector_type(4)));
typedef int          v8i __attribute__((ext_vector_type(8)));
typedef int          v4i __attribute__((ext_vector_type(4)));

#define T_TOK 2048
#define H_DIM 2048
#define E_NUM 64
#define I_DIM 512
#define TOPK  6

#if __has_builtin(__builtin_amdgcn_tensor_load_to_lds)
#define USE_TDM 1
#else
#define USE_TDM 0
#endif

union FragA { v16bf v; uint4 q[2]; };

#if USE_TDM
// Issue one TDM 2D-tile DMA: rows x cols fp32 tile (row stride in elements)
// from global into LDS at byte offset lds_byte_off. One call per wave issues
// one DMA; caller must ensure only a single wave executes this.
__device__ __forceinline__ void tdm_load_tile_f32(unsigned lds_byte_off,
                                                  const float* gsrc,
                                                  unsigned rows, unsigned cols,
                                                  unsigned row_stride) {
  unsigned long long ga = (unsigned long long)(size_t)gsrc;
  v4u g0;
  g0[0] = 1u;                                   // count=1 valid descriptor
  g0[1] = lds_byte_off;                         // lds_addr
  g0[2] = (unsigned)(ga & 0xffffffffu);         // global_addr[31:0]
  g0[3] = (unsigned)((ga >> 32) & 0x01ffffffu)  // global_addr[56:32]
          | 0x80000000u;                        // type=2 ("image")
  v8i g1;
  g1[0] = (int)(2u << 16);                      // wg_mask=0, data_size=2 (4B)
  g1[1] = (int)((cols & 0xffffu) << 16);        // tensor_dim0[15:0]
  g1[2] = (int)(((cols >> 16) & 0xffffu)        // tensor_dim0[31:16]
          | ((rows & 0xffffu) << 16));          // tensor_dim1[15:0]
  g1[3] = (int)(((rows >> 16) & 0xffffu)        // tensor_dim1[31:16]
          | ((cols & 0xffffu) << 16));          // tile_dim0 = cols
  g1[4] = (int)(rows & 0xffffu);                // tile_dim1 = rows, tile_dim2=0
  g1[5] = (int)row_stride;                      // tensor_dim0_stride[31:0]
  g1[6] = 0;                                    // stride hi / dim1_stride lo
  g1[7] = 0;
  v4i z4 = {0, 0, 0, 0};
#if __clang_major__ >= 23
  v8i z8 = {0, 0, 0, 0, 0, 0, 0, 0};
  __builtin_amdgcn_tensor_load_to_lds(g0, g1, z4, z4, z8, 0);
#else
  __builtin_amdgcn_tensor_load_to_lds(g0, g1, z4, z4, 0);
#endif
#if __has_builtin(__builtin_amdgcn_s_wait_tensorcnt)
  __builtin_amdgcn_s_wait_tensorcnt(0);
#else
  asm volatile("s_wait_tensorcnt 0x0" ::: "memory");
#endif
}
#endif  // USE_TDM

// ---------------- x fp32 -> bf16 ----------------
__global__ __launch_bounds__(256) void k_cvt(const float4* __restrict__ x,
                                             __bf16* __restrict__ xb, int n4) {
  int i = blockIdx.x * 256 + threadIdx.x;
  if (i >= n4) return;
  float4 v = x[i];
  __bf16* p = xb + (size_t)i * 4;
  p[0] = (__bf16)v.x; p[1] = (__bf16)v.y; p[2] = (__bf16)v.z; p[3] = (__bf16)v.w;
}

// ---------------- router logits: [T,E] = x @ gate_w^T ----------------
__global__ __launch_bounds__(256) void k_router(const float* __restrict__ x,
                                                const float* __restrict__ gw,
                                                float* __restrict__ logits) {
  int gid = blockIdx.x * 256 + threadIdx.x;   // 64 consecutive threads share a token
  int t = gid >> 6, e = gid & 63;
  const float4* xr = (const float4*)(x + (size_t)t * H_DIM);
  const float4* gr = (const float4*)(gw + (size_t)e * H_DIM);
  float acc = 0.f;
#pragma unroll 4
  for (int k = 0; k < H_DIM / 4; ++k) {
    float4 a = xr[k], b = gr[k];
    acc += a.x * b.x + a.y * b.y + a.z * b.z + a.w * b.w;
  }
  logits[(size_t)t * E_NUM + e] = acc;
}

// ---------------- softmax + top-6 + normalized weights + scatter ----------------
__global__ __launch_bounds__(256) void k_topk(const float* __restrict__ logits,
                                              const float* __restrict__ ebias,
                                              int* __restrict__ cnt,
                                              int* __restrict__ tok,
                                              float* __restrict__ wts) {
  int t = blockIdx.x * 256 + threadIdx.x;
  if (t >= T_TOK) return;
  float p[E_NUM];
  float m = -3.4e38f;
  for (int e = 0; e < E_NUM; ++e) { p[e] = logits[(size_t)t * E_NUM + e]; m = fmaxf(m, p[e]); }
  float s = 0.f;
  for (int e = 0; e < E_NUM; ++e) { p[e] = __expf(p[e] - m); s += p[e]; }
  float inv = 1.f / s;
  for (int e = 0; e < E_NUM; ++e) p[e] *= inv;   // softmax probs

  int   sel[TOPK];
  float w[TOPK];
  float wsum = 0.f;
  bool used[E_NUM];
  for (int e = 0; e < E_NUM; ++e) used[e] = false;
  for (int k = 0; k < TOPK; ++k) {
    float best = -3.4e38f; int bi = 0;
    for (int e = 0; e < E_NUM; ++e) {
      float sc = p[e] + ebias[e];
      if (!used[e] && sc > best) { best = sc; bi = e; }
    }
    used[bi] = true; sel[k] = bi; w[k] = p[bi]; wsum += p[bi];
  }
  wsum = fmaxf(wsum, 1e-12f);
  for (int k = 0; k < TOPK; ++k) {
    int e = sel[k];
    int slot = atomicAdd(&cnt[e], 1);
    tok[e * T_TOK + slot] = t;
    wts[e * T_TOK + slot] = w[k] / wsum;
  }
}

// ---------------- fused gate+up+SiLU per (expert, 32-col I-chunk) ----------------
// LDS: Wg,Wu chunks [32 cols][2048 K] bf16, column-major -> 2 * 128KB.
// 4 waves: wave = {ntile(0..1), m-parity(0..1)}; each wave does gate AND up WMMA.
__global__ __launch_bounds__(128) void k_gateup(const float* __restrict__ gp,
                                                const float* __restrict__ upj,
                                                const __bf16* __restrict__ xb,
                                                const int* __restrict__ cnt,
                                                const int* __restrict__ tok,
                                                __bf16* __restrict__ hbuf) {
  extern __shared__ char smem[];
  __bf16* lWg = (__bf16*)smem;          // lWg[c*2048 + k]
  __bf16* lWu = lWg + 32 * 2048;

  const int e  = blockIdx.y;
  const int i0 = blockIdx.x * 32;
  const int tid = threadIdx.x;

  const float* g0 = gp  + (size_t)e * H_DIM * I_DIM;
  const float* u0 = upj + (size_t)e * H_DIM * I_DIM;
  // stage fp32 -> bf16, transposed (column-major) into LDS
  for (int idx = tid; idx < 2048 * 8; idx += 128) {
    int k  = idx >> 3;
    int c4 = (idx & 7) << 2;
    // prefetch next K-slab toward L2 while converting this one
    __builtin_prefetch(g0 + (size_t)(k + 128) * I_DIM + i0 + c4, 0, 1);
    __builtin_prefetch(u0 + (size_t)(k + 128) * I_DIM + i0 + c4, 0, 1);
    float4 vg = *(const float4*)(g0 + (size_t)k * I_DIM + i0 + c4);
    float4 vu = *(const float4*)(u0 + (size_t)k * I_DIM + i0 + c4);
    lWg[(c4 + 0) * 2048 + k] = (__bf16)vg.x;
    lWg[(c4 + 1) * 2048 + k] = (__bf16)vg.y;
    lWg[(c4 + 2) * 2048 + k] = (__bf16)vg.z;
    lWg[(c4 + 3) * 2048 + k] = (__bf16)vg.w;
    lWu[(c4 + 0) * 2048 + k] = (__bf16)vu.x;
    lWu[(c4 + 1) * 2048 + k] = (__bf16)vu.y;
    lWu[(c4 + 2) * 2048 + k] = (__bf16)vu.z;
    lWu[(c4 + 3) * 2048 + k] = (__bf16)vu.w;
  }
  __syncthreads();

  const int wave  = tid >> 5;
  const int lane  = tid & 31;
  const int ntile = wave & 1;
  const int mpar  = wave >> 1;
  const int ce     = cnt[e];
  const int mtiles = (ce + 15) >> 4;
  const int nloc   = ntile * 16 + (lane & 15);      // B col in chunk
  const int kg     = (lane >> 4) << 3;              // dual K-group select
  const int* tokE  = tok + e * T_TOK;

  for (int mt = mpar; mt < mtiles; mt += 2) {
    const int m0  = mt << 4;
    int row  = m0 + (lane & 15);
    int ridx = row < ce ? row : ce - 1;
    const __bf16* arow = xb + (size_t)tokE[ridx] * H_DIM;

    v8f accG = {}, accU = {};
    for (int k0 = 0; k0 < H_DIM; k0 += 32) {
      FragA a, bg, bu;
      a.q[0]  = *(const uint4*)(arow + k0 + kg);
      a.q[1]  = *(const uint4*)(arow + k0 + kg + 16);
      bg.q[0] = *(const uint4*)(lWg + nloc * 2048 + k0 + kg);
      bg.q[1] = *(const uint4*)(lWg + nloc * 2048 + k0 + kg + 16);
      bu.q[0] = *(const uint4*)(lWu + nloc * 2048 + k0 + kg);
      bu.q[1] = *(const uint4*)(lWu + nloc * 2048 + k0 + kg + 16);
      accG = __builtin_amdgcn_wmma_f32_16x16x32_bf16(false, a.v, false, bg.v,
                                                     (short)0, accG, false, false);
      accU = __builtin_amdgcn_wmma_f32_16x16x32_bf16(false, a.v, false, bu.v,
                                                     (short)0, accU, false, false);
    }
    // h = silu(g) * u -> hbuf (bf16). C/D layout: VGPR r: M = r + 8*lane[4], N = lane&15
    const int N  = lane & 15;
    const int Mb = (lane >> 4) << 3;
#pragma unroll
    for (int r = 0; r < 8; ++r) {
      int orow = m0 + Mb + r;
      if (orow < ce) {
        float g = accG[r], u = accU[r];
        float h = (g / (1.f + __expf(-g))) * u;
        hbuf[((size_t)e * T_TOK + orow) * I_DIM + i0 + ntile * 16 + N] = (__bf16)h;
      }
    }
  }
}

// ---------------- down-proj per (expert, 128-col H-chunk), atomic combine ----------------
// LDS: Wd chunk [128 cols][512 K] bf16 = 128KB + 32KB TDM staging.
// 8 waves, one 16-col N-tile each. Weights staged by the Tensor Data Mover.
__global__ __launch_bounds__(256) void k_down(const float* __restrict__ dpj,
                                              const __bf16* __restrict__ hbuf,
                                              const int* __restrict__ cnt,
                                              const int* __restrict__ tok,
                                              const float* __restrict__ wts,
                                              float* __restrict__ out) {
  extern __shared__ char smem[];
  __bf16* lWd = (__bf16*)smem;          // lWd[c*512 + k], 128KB
  const int e  = blockIdx.y;
  const int h0 = blockIdx.x * 128;
  const int tid = threadIdx.x;
  const float* d0 = dpj + (size_t)e * I_DIM * H_DIM;

#if USE_TDM
  // TDM-staged: DMA 64-row x 128-col fp32 chunks (32KB) into LDS staging,
  // then cooperatively convert to bf16 column-major.
  float* stage = (float*)(smem + 128 * 1024);
  const unsigned STAGE_OFF = 128u * 1024u;   // dynamic LDS starts at offset 0
  for (int k0 = 0; k0 < I_DIM; k0 += 64) {
    if (tid < 32) {   // single wave issues the DMA and waits TENSORcnt
      tdm_load_tile_f32(STAGE_OFF, d0 + (size_t)k0 * H_DIM + h0,
                        /*rows=*/64, /*cols=*/128, /*row_stride=*/H_DIM);
    }
    __syncthreads();  // staging chunk visible to all waves
    for (int idx = tid; idx < 64 * 128; idx += 256) {
      int r = idx >> 7, c = idx & 127;
      lWd[c * 512 + (k0 + r)] = (__bf16)stage[idx];
    }
    __syncthreads();  // all reads done before next DMA overwrites staging
  }
#else
  for (int idx = tid; idx < 512 * 32; idx += 256) {
    int k  = idx >> 5;
    int c4 = (idx & 31) << 2;
    float4 v = *(const float4*)(d0 + (size_t)k * H_DIM + h0 + c4);
    lWd[(c4 + 0) * 512 + k] = (__bf16)v.x;
    lWd[(c4 + 1) * 512 + k] = (__bf16)v.y;
    lWd[(c4 + 2) * 512 + k] = (__bf16)v.z;
    lWd[(c4 + 3) * 512 + k] = (__bf16)v.w;
  }
  __syncthreads();
#endif

  const int wave = tid >> 5;
  const int lane = tid & 31;
  const int ce     = cnt[e];
  const int mtiles = (ce + 15) >> 4;
  const int ncol   = wave * 16 + (lane & 15);
  const int kg     = (lane >> 4) << 3;
  const int* tokE  = tok + e * T_TOK;
  const float* wE  = wts + e * T_TOK;

  for (int mt = 0; mt < mtiles; ++mt) {
    const int m0 = mt << 4;
    int row  = m0 + (lane & 15);
    int ridx = row < ce ? row : ce - 1;
    const __bf16* arow = hbuf + ((size_t)e * T_TOK + ridx) * I_DIM;

    v8f acc = {};
    for (int k0 = 0; k0 < I_DIM; k0 += 32) {
      FragA a, b;
      a.q[0] = *(const uint4*)(arow + k0 + kg);
      a.q[1] = *(const uint4*)(arow + k0 + kg + 16);
      b.q[0] = *(const uint4*)(lWd + ncol * 512 + k0 + kg);
      b.q[1] = *(const uint4*)(lWd + ncol * 512 + k0 + kg + 16);
      acc = __builtin_amdgcn_wmma_f32_16x16x32_bf16(false, a.v, false, b.v,
                                                    (short)0, acc, false, false);
    }
    const int N  = lane & 15;
    const int Mb = (lane >> 4) << 3;
#pragma unroll
    for (int r = 0; r < 8; ++r) {
      int orow = m0 + Mb + r;
      if (orow < ce) {
        int   t = tokE[orow];
        float w = wE[orow];
        atomicAdd(out + (size_t)t * H_DIM + h0 + wave * 16 + N, acc[r] * w);
      }
    }
  }
}

// ---------------------------------------------------------------------------
extern "C" void kernel_launch(void* const* d_in, const int* in_sizes, int n_in,
                              void* d_out, int out_size, void* d_ws, size_t ws_size,
                              hipStream_t stream) {
  const float* x   = (const float*)d_in[0];   // [2,1024,2048]
  const float* gw  = (const float*)d_in[1];   // [64,2048]
  const float* eb  = (const float*)d_in[2];   // [64]
  const float* gp  = (const float*)d_in[3];   // [64,2048,512]
  const float* up  = (const float*)d_in[4];   // [64,2048,512]
  const float* dp  = (const float*)d_in[5];   // [64,512,2048]

  float* out    = (float*)d_out;                         // final [T*H]
  float* logits = out + (size_t)T_TOK * H_DIM;           // router_logits [T*E]

  char* ws = (char*)d_ws;
  size_t off = 0;
  __bf16* xb = (__bf16*)(ws + off);  off += (size_t)T_TOK * H_DIM * 2;        // 8 MB
  int*    cnt = (int*)(ws + off);    off += 1024;
  int*    tok = (int*)(ws + off);    off += (size_t)E_NUM * T_TOK * 4;        // 512 KB
  float*  wts = (float*)(ws + off);  off += (size_t)E_NUM * T_TOK * 4;        // 512 KB
  __bf16* hbuf = (__bf16*)(ws + off);                                         // 128 MB cap

  hipMemsetAsync(out, 0, (size_t)T_TOK * H_DIM * sizeof(float), stream);
  hipMemsetAsync(cnt, 0, E_NUM * sizeof(int), stream);

  const int n4 = T_TOK * H_DIM / 4;
  k_cvt   <<<dim3((n4 + 255) / 256), 256, 0, stream>>>((const float4*)x, xb, n4);
  k_router<<<dim3(T_TOK * E_NUM / 256), 256, 0, stream>>>(x, gw, logits);
  k_topk  <<<dim3(T_TOK / 256), 256, 0, stream>>>(logits, eb, cnt, tok, wts);
  k_gateup<<<dim3(16, 64), 128, 256 * 1024, stream>>>(gp, up, xb, cnt, tok, hbuf);
  k_down  <<<dim3(16, 64), 256, 160 * 1024, stream>>>(dp, hbuf, cnt, tok, wts, out);
}
